// AttentionModel_39848706572693
// MI455X (gfx1250) — matrix operand order repaired and checked by reading, not compiled, not run on hardware
//
#include <hip/hip_runtime.h>

typedef __attribute__((ext_vector_type(16))) _Float16 v16h;
typedef __attribute__((ext_vector_type(8)))  _Float16 v8h;
typedef __attribute__((ext_vector_type(8)))  float    v8f;
typedef __attribute__((ext_vector_type(4)))  unsigned v4u;
typedef __attribute__((ext_vector_type(8)))  int      v8i;
typedef __attribute__((ext_vector_type(4)))  int      v4i;

#define B_   4
#define S_   2048
#define H_   16
#define D_   128
#define LOG2E 1.4426950408889634f
#define INV_KEEP (1.0f/0.9f)
#define KEEP_THRESH 3865470566u   /* floor(0.9 * 2^32) */

// ---------------------------------------------------------------------------
// helpers
// ---------------------------------------------------------------------------
static __device__ __forceinline__ v8f wmma_f16f32(v16h a, v16h b, v8f c) {
  return __builtin_amdgcn_wmma_f32_16x16x32_f16(
      /*neg_a=*/false, a, /*neg_b=*/false, b,
      /*c_mod=*/(short)0, c, /*reuse_a=*/false, /*reuse_b=*/false);
}

// fragment from two 16-byte-aligned 8x f16 runs (elements 0-7 <- p0, 8-15 <- p1)
static __device__ __forceinline__ v16h ldfrag(const _Float16* p0, const _Float16* p1) {
  v8h a = *(const v8h*)p0;
  v8h b = *(const v8h*)p1;
  return __builtin_shufflevector(a, b, 0,1,2,3,4,5,6,7,8,9,10,11,12,13,14,15);
}

// fragment built from f32 source (projection path): cvt on the fly
static __device__ __forceinline__ v16h cvtfrag(const float* p0, const float* p1) {
  v16h r;
#pragma unroll
  for (int i = 0; i < 8; ++i) {
    r[i]     = (_Float16)p0[i];
    r[i + 8] = (_Float16)p1[i];
  }
  return r;
}

static __device__ __forceinline__ unsigned hashu(unsigned x) {
  x ^= x >> 16; x *= 0x7feb352du;
  x ^= x >> 15; x *= 0x846ca68bu;
  x ^= x >> 16;
  return x;
}

// Low 32 bits of a generic pointer into LDS == byte offset within the
// workgroup's LDS allocation (flat-LDS aperture keeps the offset in [31:0]).
static __device__ __forceinline__ unsigned lds_off(const void* p) {
  return (unsigned)(unsigned long long)(uintptr_t)p;
}

// TDM: async-DMA a 2D tile (tile_d0 x tile_d1 elements, 2-byte data) from
// global memory into LDS, compacted row-after-row. D# per ISA 08 section 8.
static __device__ __forceinline__ void tdm_load_2d(
    unsigned ldsByteOff, unsigned long long gaddr,
    unsigned tensor_d0, unsigned tensor_d1,
    unsigned tile_d0, unsigned tile_d1, unsigned stride0)
{
  v4u g0;
  g0[0] = 1u;                                        // count=1, user descriptor
  g0[1] = ldsByteOff;                                // lds_addr
  g0[2] = (unsigned)(gaddr & 0xFFFFFFFFull);         // global_addr[31:0]
  g0[3] = (unsigned)((gaddr >> 32) & 0x1FFFFFFull)   // global_addr[56:32]
        | (2u << 30);                                // type=2 ("image")
  v8i g1;
  g1[0] = (int)0x00010000u;                          // data_size=1 -> 2 bytes
  g1[1] = (int)((tensor_d0 & 0xFFFFu) << 16);        // tensor_dim0[15:0]
  g1[2] = (int)((tensor_d0 >> 16) | ((tensor_d1 & 0xFFFFu) << 16));
  g1[3] = (int)((tensor_d1 >> 16) | (tile_d0 << 16)); // tile_dim0
  g1[4] = (int)(tile_d1 & 0xFFFFu);                  // tile_dim1 (tile_dim2=0)
  g1[5] = (int)stride0;                              // tensor_dim0_stride[31:0]
  g1[6] = 0;
  g1[7] = 0;
  v4i z4 = {0, 0, 0, 0};
#if __clang_major__ >= 23
  v8i z8 = {0, 0, 0, 0, 0, 0, 0, 0};
  __builtin_amdgcn_tensor_load_to_lds(g0, g1, z4, z4, z8, 0);
#else
  __builtin_amdgcn_tensor_load_to_lds(g0, g1, z4, z4, 0);
#endif
}

// ---------------------------------------------------------------------------
// Kernel 1: fused linear projection  out = (X @ W^T + b) * postScale  (-> f16)
//   X: [B*S*H, 128] f32 rows;  W: [128,128];  mode 0/1: out [B,H,S,D]
//   mode 2: out [B,H,D,S]  (V stored transposed for the PV WMMA B-layout)
// Each block: 16 rows x 128 cols; each of 8 waves owns one 16x16 tile.
// ---------------------------------------------------------------------------
__global__ __launch_bounds__(256) void proj_kernel(
    const float* __restrict__ X, const float* __restrict__ W,
    const float* __restrict__ bias, _Float16* __restrict__ out,
    int mode, float postScale)
{
  const int lane   = threadIdx.x & 31;
  const int wave   = threadIdx.x >> 5;
  const int laneLo = lane & 15;
  const int hi     = lane >> 4;          // 0 or 1 (half-wave)
  const int rBase  = blockIdx.x * 16;    // 16 consecutive rows == 16 heads of one (b,s)
  const int jBase  = wave * 16;          // output column tile

  v8f acc = {};
#pragma unroll
  for (int kc = 0; kc < 4; ++kc) {
    // A: 16x32 tile of X (rows = (b,s,h), K = input dim)
    const float* xp = X + (size_t)(rBase + laneLo) * D_ + kc * 32 + hi * 8;
    v16h a = cvtfrag(xp, xp + 16);
    // B: 32x16 of W^T -> column n is row (jBase+n) of W, contiguous along d
    const float* wp = W + (size_t)(jBase + laneLo) * D_ + kc * 32 + hi * 8;
    v16h b = cvtfrag(wp, wp + 16);
    acc = wmma_f16f32(a, b, acc);
  }

  const float bv = bias[jBase + laneLo];
  const int bs = rBase >> 4;       // = b*S + s  (H_ == 16)
  const int bb = bs / S_;
  const int ss = bs % S_;
#pragma unroll
  for (int i = 0; i < 8; ++i) {
    const int m = i + hi * 8;      // row in tile == head index h
    const float val = (acc[i] + bv) * postScale;
    size_t addr;
    if (mode == 2) {  // [B,H,D,S]
      addr = (((size_t)(bb * H_ + m)) * D_ + (jBase + laneLo)) * S_ + ss;
    } else {          // [B,H,S,D]
      addr = (((size_t)(bb * H_ + m)) * S_ + ss) * D_ + (jBase + laneLo);
    }
    out[addr] = (_Float16)val;
  }
}

// ---------------------------------------------------------------------------
// Kernel 2: flash attention. Block = 8 waves x 16 queries = 128 queries.
// K/V chunks (shared by all 8 waves) are staged into LDS once per block by
// the Tensor Data Mover, double-buffered to overlap DMA with WMMA compute.
// ---------------------------------------------------------------------------
__global__ __launch_bounds__(256) void attn_kernel(
    const _Float16* __restrict__ Qp,  // [BH,S,D] f16, pre-scaled by 1/sqrt(D)
    const _Float16* __restrict__ Kp,  // [BH,S,D] f16
    const _Float16* __restrict__ Vt,  // [BH,D,S] f16 (transposed)
    float* __restrict__ out)          // [BH,S,D] f32
{
  __shared__ __align__(16) _Float16 Kbuf[2][32 * 128];  // 2 x 8KB
  __shared__ __align__(16) _Float16 Vbuf[2][128 * 32];  // 2 x 8KB
  __shared__ __align__(16) _Float16 Pbuf[8][512];       // 1KB / wave

  const int lane   = threadIdx.x & 31;
  const int wave   = threadIdx.x >> 5;
  const int laneLo = lane & 15;
  const int hi     = lane >> 4;
  const int bh     = blockIdx.y;
  const int qBase  = blockIdx.x * 128 + wave * 16;

  const _Float16* Q = Qp + (size_t)bh * S_ * D_;
  const _Float16* K = Kp + (size_t)bh * S_ * D_;
  const _Float16* V = Vt + (size_t)bh * D_ * S_;
  _Float16* pbuf = Pbuf[wave];

  // provably-uniform leader predicate -> scalar branch (TDM ignores EXEC,
  // so an EXEC-masked guard would mis-issue from every wave)
  const bool leader = (__builtin_amdgcn_readfirstlane((int)(threadIdx.x >> 5)) == 0);

  // Resident Q A-fragments (16 queries x 128 d) = 4 fragments of 16x32
  v16h aq[4];
#pragma unroll
  for (int dc = 0; dc < 4; ++dc) {
    const _Float16* p = Q + (size_t)(qBase + laneLo) * D_ + dc * 32 + hi * 8;
    aq[dc] = ldfrag(p, p + 16);
  }

  v8f o[8];
#pragma unroll
  for (int n = 0; n < 8; ++n) o[n] = (v8f){};
  float mrow[8], lrow[8];
#pragma unroll
  for (int i = 0; i < 8; ++i) { mrow[i] = -1e30f; lrow[i] = 0.0f; }

  // prologue: DMA chunk 0 into buffer 0
  if (leader) {
    tdm_load_2d(lds_off(Kbuf[0]), (unsigned long long)(uintptr_t)K,
                /*tensor_d0=*/D_, /*tensor_d1=*/S_,
                /*tile_d0=*/D_, /*tile_d1=*/32, /*stride0=*/D_);
    tdm_load_2d(lds_off(Vbuf[0]), (unsigned long long)(uintptr_t)V,
                /*tensor_d0=*/S_, /*tensor_d1=*/D_,
                /*tile_d0=*/32, /*tile_d1=*/D_, /*stride0=*/S_);
  }

  for (int kc = 0; kc < S_ / 32; ++kc) {
    const int cur = kc & 1;
    if (leader) __builtin_amdgcn_s_wait_tensorcnt(0);
    __syncthreads();   // buf[cur] ready for all waves; buf[cur^1] free

    // kick off next chunk's DMA while we compute on the current one
    if (leader && (kc + 1 < S_ / 32)) {
      const size_t kNext = (size_t)(kc + 1) * 32;
      tdm_load_2d(lds_off(Kbuf[cur ^ 1]),
                  (unsigned long long)(uintptr_t)(K + kNext * D_),
                  D_, S_, D_, 32, D_);
      tdm_load_2d(lds_off(Vbuf[cur ^ 1]),
                  (unsigned long long)(uintptr_t)(V + kNext),
                  S_, D_, 32, D_, S_);
    }

    const _Float16* Kc = Kbuf[cur];   // [32 keys][128 d]
    const _Float16* Vc = Vbuf[cur];   // [128 d][32 keys]

    // ---- S = Q K^T for 32 keys (two 16-key n-tiles), K-loop over D
    v8f s0 = {}, s1 = {};
#pragma unroll
    for (int dc = 0; dc < 4; ++dc) {
      const _Float16* kp0 = Kc + (laneLo) * D_ + dc * 32 + hi * 8;
      v16h b0 = ldfrag(kp0, kp0 + 16);
      s0 = wmma_f16f32(aq[dc], b0, s0);
      const _Float16* kp1 = Kc + (16 + laneLo) * D_ + dc * 32 + hi * 8;
      v16h b1 = ldfrag(kp1, kp1 + 16);
      s1 = wmma_f16f32(aq[dc], b1, s1);
    }

    // ---- online softmax + dropout, stage P (f16) into wave-private LDS
    const int kBase = kc * 32;
#pragma unroll
    for (int i = 0; i < 8; ++i) {
      // row of this VGPR slot for this lane: i (low half) / i+8 (high half)
      float mx = fmaxf(s0[i], s1[i]);
      mx = fmaxf(mx, __shfl_xor(mx, 1, 32));
      mx = fmaxf(mx, __shfl_xor(mx, 2, 32));
      mx = fmaxf(mx, __shfl_xor(mx, 4, 32));
      mx = fmaxf(mx, __shfl_xor(mx, 8, 32));   // reduce within 16-lane half
      const float mnew = fmaxf(mrow[i], mx);
      const float corr = __builtin_exp2f((mrow[i] - mnew) * LOG2E);
      mrow[i] = mnew;
      float p0 = __builtin_exp2f((s0[i] - mnew) * LOG2E);
      float p1 = __builtin_exp2f((s1[i] - mnew) * LOG2E);
      float ps = p0 + p1;
      ps += __shfl_xor(ps, 1, 32);
      ps += __shfl_xor(ps, 2, 32);
      ps += __shfl_xor(ps, 4, 32);
      ps += __shfl_xor(ps, 8, 32);
      lrow[i] = lrow[i] * corr + ps;
#pragma unroll
      for (int n = 0; n < 8; ++n) o[n][i] *= corr;  // rescale O (same row map)

      // deterministic inverted dropout (post-normalization in the reference;
      // commutes with the row-scalar 1/l, so apply to unnormalized P here)
      const int row = i + hi * 8;
      const unsigned q    = (unsigned)(qBase + row);
      const unsigned idx0 = ((unsigned)bh << 22) + (q << 11) + (unsigned)(kBase + laneLo);
      p0 = (hashu(idx0)      < KEEP_THRESH) ? p0 * INV_KEEP : 0.0f;
      p1 = (hashu(idx0 + 16) < KEEP_THRESH) ? p1 * INV_KEEP : 0.0f;
      pbuf[row * 32 + laneLo]      = (_Float16)p0;
      pbuf[row * 32 + 16 + laneLo] = (_Float16)p1;
    }

    // ---- reload P in A-fragment layout (16 q x 32 keys); same-wave DS order
    const _Float16* pp = pbuf + laneLo * 32 + hi * 8;
    v16h pa = ldfrag(pp, pp + 16);

    // ---- O += P @ V  (8 d-tiles of 16) from the LDS-staged V chunk
#pragma unroll
    for (int n = 0; n < 8; ++n) {
      const _Float16* vp = Vc + (n * 16 + laneLo) * 32 + hi * 8;
      v16h vb = ldfrag(vp, vp + 16);
      o[n] = wmma_f16f32(pa, vb, o[n]);
    }
  }

  // ---- finalize: O / l, write f32 [BH,S,D]
#pragma unroll
  for (int i = 0; i < 8; ++i) {
    const int row = i + hi * 8;
    const float inv = 1.0f / lrow[i];
#pragma unroll
    for (int n = 0; n < 8; ++n) {
      out[((size_t)bh * S_ + qBase + row) * D_ + n * 16 + laneLo] = o[n][i] * inv;
    }
  }
}

// ---------------------------------------------------------------------------
extern "C" void kernel_launch(void* const* d_in, const int* in_sizes, int n_in,
                              void* d_out, int out_size, void* d_ws, size_t ws_size,
                              hipStream_t stream) {
  (void)in_sizes; (void)n_in; (void)out_size; (void)ws_size;
  const float* query = (const float*)d_in[0];
  const float* key   = (const float*)d_in[1];
  const float* value = (const float*)d_in[2];
  const float* Wq    = (const float*)d_in[3];
  const float* bq    = (const float*)d_in[4];
  const float* Wk    = (const float*)d_in[5];
  const float* bk    = (const float*)d_in[6];
  const float* Wv    = (const float*)d_in[7];
  const float* bv    = (const float*)d_in[8];

  const size_t elems = (size_t)B_ * S_ * H_ * D_;   // 16,777,216
  _Float16* Qp = (_Float16*)d_ws;
  _Float16* Kp = Qp + elems;
  _Float16* Vt = Kp + elems;

  const int nblk = (B_ * S_ * H_) / 16;             // 8192 row-tiles
  const float qscale = 0.08838834764831845f;        // 1/sqrt(128), folded into Q

  hipLaunchKernelGGL(proj_kernel, dim3(nblk), dim3(256), 0, stream,
                     query, Wq, bq, Qp, 0, qscale);
  hipLaunchKernelGGL(proj_kernel, dim3(nblk), dim3(256), 0, stream,
                     key,   Wk, bk, Kp, 1, 1.0f);
  hipLaunchKernelGGL(proj_kernel, dim3(nblk), dim3(256), 0, stream,
                     value, Wv, bv, Vt, 2, 1.0f);

  hipLaunchKernelGGL(attn_kernel, dim3(S_ / 128, B_ * H_), dim3(256), 0, stream,
                     Qp, Kp, Vt, (float*)d_out);
}
